// LocalLSTM_83777632076003
// MI455X (gfx1250) — compile-verified, matrix operand order checked by reading, not run on hardware
//
#include <hip/hip_runtime.h>
#include <hip/hip_bf16.h>

// ---------------------------------------------------------------------------
// CDNA5 (gfx1250) persistent-LSTM implementation.
//   grid  = 128 workgroups x 256 threads (8 waves of 32)
//   each workgroup owns 128 nodes; each wave owns a 16-node row tile
//   c-state lives in VGPRs, h-state lives in LDS, weights pre-packed fp16
//   gates GEMMs use v_wmma_f32_16x16x32_f16 (fp32 accumulate)
// ---------------------------------------------------------------------------

typedef __attribute__((ext_vector_type(16))) _Float16 v16h;
typedef __attribute__((ext_vector_type(8)))  float    v8f;

// Global-address-space views so weight loads lower to global_load_b128
// (FLAT loads would also tick DScnt and serialize against the LDS pipe).
typedef __attribute__((address_space(1))) const _Float16 GHalf;
typedef __attribute__((address_space(1))) const v16h     Gv16h;

union AF { v16h v; uint4 q[2]; };
union HF { uint4 q; _Float16 h[8]; };

__device__ __forceinline__ float sigf(float x) { return 1.0f / (1.0f + __expf(-x)); }

// Launder the packed-weight pointer through an empty asm so the loads cannot
// be hoisted/CSE'd out of the recurrent loop (prevents the compiler from
// keeping ~8KB/lane of weight fragments live -> scratch spills), while the
// explicit addrspace(1) cast keeps them on the global-load path.
__device__ __forceinline__ const GHalf* opaque(const _Float16* p) {
    const GHalf* g = (const GHalf*)(unsigned long long)p;
    asm volatile("" : "+s"(g));
    return g;
}

// ---------------------------------------------------------------------------
// Packed-B layout (mirrors the wave32 WMMA 16-bit fragment striping):
//   tile (ct, kt) holds B[k][n] = W[ct*16 + n][kt*32 + klocal]
//   lane L (n = L&15), half i: klocal = (i<8 ? i : i+8) + ((L>=16) ? 8 : 0)
// so every lane loads its 16 halves as one contiguous 32B chunk.
// The A fragment from LDS uses the identical klocal mapping (two 16B reads).
// ---------------------------------------------------------------------------
__global__ void pack_w(_Float16* __restrict__ dst,
                       const float* __restrict__ W1, const float* __restrict__ W2,
                       int K1, int K2, int nCT) {
    const int nKT = (K1 + K2) >> 5;
    const int total = nCT * nKT * 512;
    for (int idx = blockIdx.x * blockDim.x + threadIdx.x; idx < total;
         idx += gridDim.x * blockDim.x) {
        const int i    = idx & 15;
        const int lane = (idx >> 4) & 31;
        const int t2   = idx >> 9;          // ct*nKT + kt
        const int kt   = t2 % nKT;
        const int ct   = t2 / nKT;
        const int n    = ct * 16 + (lane & 15);
        const int kl   = (i < 8 ? i : i + 8) + ((lane >= 16) ? 8 : 0);
        const int k    = kt * 32 + kl;
        const float v  = (k < K1) ? W1[n * K1 + k] : W2[n * K2 + (k - K1)];
        dst[idx] = (_Float16)v;
    }
}

__global__ void combine_bias(float* __restrict__ dst,
                             const float* __restrict__ a,
                             const float* __restrict__ b, int n) {
    const int i = blockIdx.x * blockDim.x + threadIdx.x;
    if (i < n) dst[i] = a[i] + b[i];
}

// ---------------------------------------------------------------------------
// One fused LSTM layer step for a wave's 16-node tile.
// gates = A[16 x (K1+K2)] @ Wcomb^T, then elementwise LSTM update.
// K segments come from the activation buffer at S0OFF (S0CNT k-tiles) and
// S1OFF (rest).  Gate g, hidden chunk hc -> packed col tile ct = g*8 + hc.
// New h is staged in hStage, then block-copied into the activation buffer
// (avoids overwriting A columns that later hc iterations still read).
// ---------------------------------------------------------------------------
template<int NKT, int S0OFF, int S0CNT, int S1OFF>
__device__ __forceinline__ void lstm_layer(
    const _Float16* __restrict__ actRowLane,   // LDS &act[(waveBase+(lane&15))*512]
    int laneKb,                                 // (lane>=16)?8:0
    const _Float16* __restrict__ BpkIn,         // packed weights (global, L2-resident)
    const float* __restrict__ biasLds,          // 512 combined bias (bih+bhh)
    float (&cReg)[8][8],
    _Float16* __restrict__ hStageLane,          // LDS &hStage[waveBase*128]
    _Float16* __restrict__ actHOut,             // LDS &act[waveBase*512 + HCOL]
    int lane) {
    const GHalf* Bpk = opaque(BpkIn);           // keep B loads inside the loop
    const int nloc = lane & 15;
    const int mAdd = (lane >= 16) ? 8 : 0;
#pragma unroll
    for (int hc = 0; hc < 8; ++hc) {
        v8f a0 = {}, a1 = {}, a2 = {}, a3 = {};
#pragma unroll
        for (int kt = 0; kt < NKT; ++kt) {
            const int kcol = (kt < S0CNT) ? (S0OFF + kt * 32)
                                          : (S1OFF + (kt - S0CNT) * 32);
            AF af;
            af.q[0] = *(const uint4*)(actRowLane + kcol + laneKb);
            af.q[1] = *(const uint4*)(actRowLane + kcol + laneKb + 16);
            v16h b;
            b = *(const Gv16h*)(Bpk + (size_t)((0 * 8 + hc) * NKT + kt) * 512 + lane * 16);
            a0 = __builtin_amdgcn_wmma_f32_16x16x32_f16(false, af.v, false, b, (short)0, a0, false, false);
            b = *(const Gv16h*)(Bpk + (size_t)((1 * 8 + hc) * NKT + kt) * 512 + lane * 16);
            a1 = __builtin_amdgcn_wmma_f32_16x16x32_f16(false, af.v, false, b, (short)0, a1, false, false);
            b = *(const Gv16h*)(Bpk + (size_t)((2 * 8 + hc) * NKT + kt) * 512 + lane * 16);
            a2 = __builtin_amdgcn_wmma_f32_16x16x32_f16(false, af.v, false, b, (short)0, a2, false, false);
            b = *(const Gv16h*)(Bpk + (size_t)((3 * 8 + hc) * NKT + kt) * 512 + lane * 16);
            a3 = __builtin_amdgcn_wmma_f32_16x16x32_f16(false, af.v, false, b, (short)0, a3, false, false);
        }
        const float bi = biasLds[      hc * 16 + nloc];
        const float bf = biasLds[128 + hc * 16 + nloc];
        const float bg = biasLds[256 + hc * 16 + nloc];
        const float bo = biasLds[384 + hc * 16 + nloc];
#pragma unroll
        for (int j = 0; j < 8; ++j) {
            const float iv = sigf(a0[j] + bi);
            const float fv = sigf(a1[j] + bf);
            const float gv = tanhf(a2[j] + bg);
            const float ov = sigf(a3[j] + bo);
            const float cn = fv * cReg[hc][j] + iv * gv;
            cReg[hc][j] = cn;
            const float hv = ov * tanhf(cn);
            hStageLane[(j + mAdd) * 128 + hc * 16 + nloc] = (_Float16)hv;
        }
    }
    // publish staged h into the activation buffer (wave-private rows)
#pragma unroll
    for (int q = 0; q < 4; ++q) {
        const int li  = lane + 32 * q;   // 0..127 over [16 rows][8 uint4]
        const int row = li >> 3;
        const int qc  = li & 7;
        *(uint4*)(actHOut + row * 512 + qc * 8) =
            *(const uint4*)(hStageLane + row * 128 + qc * 8);
    }
}

// Head hidden layer: z = relu(h1 @ W1^T + b1) -> LDS cols 0..127 (fp16).
__device__ __forceinline__ void head_gemm(
    const _Float16* __restrict__ actRowLane, int laneKb,
    const _Float16* __restrict__ BpkIn,
    const float* __restrict__ b1Lds,
    _Float16* __restrict__ zOut,   // &act[waveBase*512]
    int lane) {
    const GHalf* Bpk = opaque(BpkIn);
    const int nloc = lane & 15;
    const int mAdd = (lane >= 16) ? 8 : 0;
#pragma unroll
    for (int ct = 0; ct < 8; ++ct) {
        v8f acc = {};
#pragma unroll
        for (int kt = 0; kt < 4; ++kt) {
            const int kcol = 384 + kt * 32;   // h1 lives in cols 384..511
            AF af;
            af.q[0] = *(const uint4*)(actRowLane + kcol + laneKb);
            af.q[1] = *(const uint4*)(actRowLane + kcol + laneKb + 16);
            v16h b = *(const Gv16h*)(Bpk + (size_t)(ct * 4 + kt) * 512 + lane * 16);
            acc = __builtin_amdgcn_wmma_f32_16x16x32_f16(false, af.v, false, b, (short)0, acc, false, false);
        }
        const float bb = b1Lds[ct * 16 + nloc];
#pragma unroll
        for (int j = 0; j < 8; ++j) {
            float z = acc[j] + bb;
            z = z > 0.0f ? z : 0.0f;
            zOut[(j + mAdd) * 512 + ct * 16 + nloc] = (_Float16)z;
        }
    }
}

// Vectorized H->1 reduction over one node's z row (128 fp16 in LDS).
__device__ __forceinline__ float head_reduce(const _Float16* __restrict__ zRow,
                                             const float* __restrict__ w2,
                                             float bias) {
    float s = bias;
#pragma unroll
    for (int q = 0; q < 16; ++q) {
        HF hv;
        hv.q = *(const uint4*)(zRow + q * 8);
#pragma unroll
        for (int j = 0; j < 8; ++j) s += (float)hv.h[j] * w2[q * 8 + j];
    }
    return s;
}

// ---------------------------------------------------------------------------
// Persistent main kernel.
// Activation buffer columns (fp16): [0:128)=inp/z-scratch  [128:256)=enc_state
//                                   [256:384)=h0           [384:512)=h1
// ---------------------------------------------------------------------------
__global__ __launch_bounds__(256, 1) void lstm_main(
    const float* __restrict__ x, const float* __restrict__ coords,
    const float* __restrict__ env,
    const float* __restrict__ encWin, const float* __restrict__ nodeWin,
    const _Float16* __restrict__ pkEnc0, const _Float16* __restrict__ pkEnc1,
    const _Float16* __restrict__ pkIn,   const _Float16* __restrict__ pkL1,
    const _Float16* __restrict__ pkH2in, const _Float16* __restrict__ pkH2out,
    const float* __restrict__ bEnc0, const float* __restrict__ bEnc1,
    const float* __restrict__ bIn,   const float* __restrict__ bL1,
    const float* __restrict__ h2inB1,  const float* __restrict__ h2inW2,  const float* __restrict__ h2inB2,
    const float* __restrict__ h2outB1, const float* __restrict__ h2outW2, const float* __restrict__ h2outB2,
    float* __restrict__ out) {
    __shared__ __align__(16) _Float16 act[128 * 512];
    __shared__ __align__(16) _Float16 hStage[128 * 128];
    __shared__ float invec[128 * 10];
    __shared__ float sWinE[1280];
    __shared__ float sWinN[1280];
    __shared__ float sB0[512], sB1[512], sBin[512], sBl1[512];
    __shared__ float sH1b1[128], sH1W2[128], sH2b1[128], sH2W2[128];
    __shared__ float sXc[128], sXin[128];
    __shared__ float sB2[2];

    const int tid      = threadIdx.x;
    const int lane     = tid & 31;
    const int wave     = tid >> 5;
    const int nodeBase = blockIdx.x * 128;          // global first node
    const int wb       = wave * 16;                 // wave's first local node

    // persistent small params -> LDS
    for (int i = tid; i < 1280; i += 256) { sWinE[i] = encWin[i]; sWinN[i] = nodeWin[i]; }
    for (int i = tid; i < 512; i += 256) {
        sB0[i] = bEnc0[i]; sB1[i] = bEnc1[i]; sBin[i] = bIn[i]; sBl1[i] = bL1[i];
    }
    if (tid < 128) {
        sH1b1[tid] = h2inB1[tid];  sH1W2[tid] = h2inW2[tid];
        sH2b1[tid] = h2outB1[tid]; sH2W2[tid] = h2outW2[tid];
    }
    if (tid == 0) { sB2[0] = h2inB2[0]; sB2[1] = h2outB2[0]; }
    for (int i = tid; i < 128 * 512; i += 256) act[i] = (_Float16)0.0f;
    __syncthreads();

    float c0[8][8];
    float c1[8][8];
#pragma unroll
    for (int a = 0; a < 8; ++a)
#pragma unroll
        for (int b = 0; b < 8; ++b) { c0[a][b] = 0.0f; c1[a][b] = 0.0f; }

    const int laneKb = (lane >= 16) ? 8 : 0;
    const _Float16* rowp = &act[(size_t)(wb + (lane & 15)) * 512];
    _Float16* hStg  = &hStage[(size_t)wb * 128];
    _Float16* h0Out = &act[(size_t)wb * 512 + 256];
    _Float16* h1Out = &act[(size_t)wb * 512 + 384];
    _Float16* zOut  = &act[(size_t)wb * 512];

    // ------------------------------ encoder ------------------------------
    for (int t = 0; t < 24; ++t) {
        for (int i = tid; i < 1280; i += 256) {
            const int node = i / 10, k = i - node * 10;
            const int gn = nodeBase + node;
            float v;
            if (k == 0)      v = x[(size_t)gn * 64 + t];
            else if (k < 3)  v = coords[(size_t)gn * 2 + (k - 1)];
            else             v = env[((size_t)gn * 7 + (k - 3)) * 64 + t];
            invec[i] = v;
        }
        __syncthreads();
        for (int i = tid; i < 128 * 128; i += 256) {     // inp = invec @ Win^T
            const int node = i >> 7, f = i & 127;
            const float* iv = &invec[node * 10];
            const float* wr = &sWinE[f * 10];
            float s = 0.0f;
#pragma unroll
            for (int k = 0; k < 10; ++k) s += iv[k] * wr[k];
            act[(node << 9) + f] = (_Float16)s;
        }
        __syncthreads();
        // layer0: A = [inp(0:128) , h0(256:384)], K=256
        lstm_layer<8, 0, 4, 256>(rowp, laneKb, pkEnc0, sB0, c0, hStg, h0Out, lane);
        // layer1: A = [h0(256:384), h1(384:512)], K=256
        lstm_layer<8, 256, 8, 0>(rowp, laneKb, pkEnc1, sB1, c1, hStg, h1Out, lane);
        __syncthreads();
    }

    // enc_state = final h1 -> cols 128:256 ; xc = x[:, 23]
    for (int i = tid; i < 128 * 128; i += 256) {
        const int node = i >> 7, f = i & 127;
        act[node * 512 + 128 + f] = act[node * 512 + 384 + f];
    }
    if (tid < 128) sXc[tid] = x[(size_t)(nodeBase + tid) * 64 + 23];
    __syncthreads();

    // ------------------------------ decoder ------------------------------
    for (int t = 0; t < 40; ++t) {
        for (int i = tid; i < 1280; i += 256) {
            const int node = i / 10, k = i - node * 10;
            const int gn = nodeBase + node;
            float v;
            if (k == 0)      v = sXc[node];
            else if (k < 3)  v = coords[(size_t)gn * 2 + (k - 1)];
            else             v = env[((size_t)gn * 7 + (k - 3)) * 64 + 24 + t];
            invec[i] = v;
        }
        __syncthreads();
        for (int i = tid; i < 128 * 128; i += 256) {
            const int node = i >> 7, f = i & 127;
            const float* iv = &invec[node * 10];
            const float* wr = &sWinN[f * 10];
            float s = 0.0f;
#pragma unroll
            for (int k = 0; k < 10; ++k) s += iv[k] * wr[k];
            act[(node << 9) + f] = (_Float16)s;
        }
        __syncthreads();
        // lstm_in: A = [inp | enc_state | h0] = cols 0..383, K=384
        lstm_layer<12, 0, 12, 0>(rowp, laneKb, pkIn, sBin, c0, hStg, h0Out, lane);
        // l1: A = [h0 | h1] = cols 256..511, K=256
        lstm_layer<8, 256, 8, 0>(rowp, laneKb, pkL1, sBl1, c1, hStg, h1Out, lane);
        __syncthreads();

        head_gemm(rowp, laneKb, pkH2in, sH1b1, zOut, lane);
        __syncthreads();
        if (tid < 128) {
            sXin[tid] = sigf(head_reduce(&act[tid * 512], sH1W2, sB2[0]));
        }
        __syncthreads();
        head_gemm(rowp, laneKb, pkH2out, sH2b1, zOut, lane);
        __syncthreads();
        if (tid < 128) {
            const float xout = sigf(head_reduce(&act[tid * 512], sH2W2, sB2[1]));
            const float xc = sXc[tid];
            const float xn = xc + sXin[tid] - xout * xc;
            sXc[tid] = xn;
            out[(size_t)(nodeBase + tid) * 40 + t] = xn;
        }
        __syncthreads();
    }
}

// ---------------------------------------------------------------------------
// workspace layout (bytes)
// ---------------------------------------------------------------------------
static const size_t OFF_ENC0  = 0;            // 32ct x  8kt : 262144 B
static const size_t OFF_ENC1  = 262144;       // 262144 B
static const size_t OFF_IN    = 524288;       // 32ct x 12kt : 393216 B
static const size_t OFF_L1    = 917504;       // 262144 B
static const size_t OFF_H2IN  = 1179648;      //  8ct x  4kt : 32768 B
static const size_t OFF_H2OUT = 1212416;      // 32768 B
static const size_t OFF_B0    = 1245184;      // 4 x 2048 B fp32 combined biases
static const size_t OFF_B1    = 1247232;
static const size_t OFF_BIN   = 1249280;
static const size_t OFF_BL1   = 1251328;

extern "C" void kernel_launch(void* const* d_in, const int* in_sizes, int n_in,
                              void* d_out, int out_size, void* d_ws, size_t ws_size,
                              hipStream_t stream) {
    const float* x       = (const float*)d_in[0];
    const float* coords  = (const float*)d_in[1];
    const float* env     = (const float*)d_in[2];
    const float* encWin  = (const float*)d_in[3];
    const float* encWih0 = (const float*)d_in[4];
    const float* encWhh0 = (const float*)d_in[5];
    const float* encBih0 = (const float*)d_in[6];
    const float* encBhh0 = (const float*)d_in[7];
    const float* encWih1 = (const float*)d_in[8];
    const float* encWhh1 = (const float*)d_in[9];
    const float* encBih1 = (const float*)d_in[10];
    const float* encBhh1 = (const float*)d_in[11];
    const float* nodeWin = (const float*)d_in[12];
    const float* inWih   = (const float*)d_in[13];
    const float* inWhh   = (const float*)d_in[14];
    const float* inBih   = (const float*)d_in[15];
    const float* inBhh   = (const float*)d_in[16];
    const float* l1Wih   = (const float*)d_in[17];
    const float* l1Whh   = (const float*)d_in[18];
    const float* l1Bih   = (const float*)d_in[19];
    const float* l1Bhh   = (const float*)d_in[20];
    const float* h2inW1  = (const float*)d_in[21];
    const float* h2inB1  = (const float*)d_in[22];
    const float* h2inW2  = (const float*)d_in[23];
    const float* h2inB2  = (const float*)d_in[24];
    const float* h2outW1 = (const float*)d_in[25];
    const float* h2outB1 = (const float*)d_in[26];
    const float* h2outW2 = (const float*)d_in[27];
    const float* h2outB2 = (const float*)d_in[28];

    char* ws = (char*)d_ws;
    _Float16* pkEnc0  = (_Float16*)(ws + OFF_ENC0);
    _Float16* pkEnc1  = (_Float16*)(ws + OFF_ENC1);
    _Float16* pkIn    = (_Float16*)(ws + OFF_IN);
    _Float16* pkL1    = (_Float16*)(ws + OFF_L1);
    _Float16* pkH2in  = (_Float16*)(ws + OFF_H2IN);
    _Float16* pkH2out = (_Float16*)(ws + OFF_H2OUT);
    float* bEnc0 = (float*)(ws + OFF_B0);
    float* bEnc1 = (float*)(ws + OFF_B1);
    float* bIn   = (float*)(ws + OFF_BIN);
    float* bL1   = (float*)(ws + OFF_BL1);

    // prep: pack weights (combined [Wih | Whh]) into WMMA-B fragment order
    pack_w<<<128, 256, 0, stream>>>(pkEnc0, encWih0, encWhh0, 128, 128, 32);
    pack_w<<<128, 256, 0, stream>>>(pkEnc1, encWih1, encWhh1, 128, 128, 32);
    pack_w<<<128, 256, 0, stream>>>(pkIn,   inWih,   inWhh,   256, 128, 32);
    pack_w<<<128, 256, 0, stream>>>(pkL1,   l1Wih,   l1Whh,   128, 128, 32);
    pack_w<<<16,  256, 0, stream>>>(pkH2in,  h2inW1,  h2inW1,  128, 0, 8);
    pack_w<<<16,  256, 0, stream>>>(pkH2out, h2outW1, h2outW1, 128, 0, 8);
    combine_bias<<<2, 256, 0, stream>>>(bEnc0, encBih0, encBhh0, 512);
    combine_bias<<<2, 256, 0, stream>>>(bEnc1, encBih1, encBhh1, 512);
    combine_bias<<<2, 256, 0, stream>>>(bIn,   inBih,   inBhh,   512);
    combine_bias<<<2, 256, 0, stream>>>(bL1,   l1Bih,   l1Bhh,   512);

    // persistent recurrent kernel: 16384 nodes / 128 per workgroup
    lstm_main<<<128, 256, 0, stream>>>(
        x, coords, env, encWin, nodeWin,
        pkEnc0, pkEnc1, pkIn, pkL1, pkH2in, pkH2out,
        bEnc0, bEnc1, bIn, bL1,
        h2inB1, h2inW2, h2inB2, h2outB1, h2outW2, h2outB2,
        (float*)d_out);
}
